// CausalMultiHeadSelfAttention_317827580052
// MI455X (gfx1250) — matrix operand order repaired and checked by reading, not compiled, not run on hardware
//
#include <hip/hip_runtime.h>
#include <hip/hip_bf16.h>

// ---------------------------------------------------------------------------
// Causal MHA + RoPE for gfx1250 (MI455X), bf16 WMMA everywhere.
// b=4, s=2048, d_model=1024, heads=16, d_k=64.
// Attention computes S^T = K·Q^T and O^T = V^T·P so online-softmax state is a
// per-lane scalar. 32 queries per wave (2 query tiles share K/V fragments).
// Diagonal (masked) key-block handled outside the hot loop.
// GEMM uses ping-pong double buffering (no register copies).
// ---------------------------------------------------------------------------

typedef __attribute__((ext_vector_type(16))) __bf16 v16bf;
typedef __attribute__((ext_vector_type(8)))  float  v8f;

union Frag16 {
    v16bf v;
    uint4 q[2];
    unsigned short u[16];
};

__device__ __forceinline__ unsigned short f2bf(float f) {
    unsigned int u = __float_as_uint(f);
    u += 0x7FFFu + ((u >> 16) & 1u);          // round-to-nearest-even
    return (unsigned short)(u >> 16);
}
__device__ __forceinline__ float bf2f(unsigned short h) {
    return __uint_as_float(((unsigned int)h) << 16);
}

// ---------------------------------------------------------------------------
// f32 -> bf16 elementwise convert
// ---------------------------------------------------------------------------
__global__ void cvt_f32_bf16(const float* __restrict__ in,
                             unsigned short* __restrict__ out, int n) {
    int i = blockIdx.x * blockDim.x + threadIdx.x;
    if (i < n) out[i] = f2bf(in[i]);
}

// ---------------------------------------------------------------------------
// GEMM: C[M,N] = A[M,K] * B[N,K]^T   (bf16 in, f32 accumulate)
// Wave tile: 32 x 64 (8 accumulators, B reused across 2 M-tiles).
// Ping-pong double-buffered fragments, K must be a multiple of 64.
// Block = 8 waves => 256 x 64 tile. grid = (M/256, N/64)
// ---------------------------------------------------------------------------
template <bool BF16OUT>
__global__ void gemm_bf16(const unsigned short* __restrict__ A,
                          const unsigned short* __restrict__ B,
                          float* __restrict__ Cf,
                          unsigned short* __restrict__ Cb,
                          int K, int ldc) {
    const int lane = threadIdx.x & 31;
    const int wave = threadIdx.x >> 5;
    const int nl   = lane & 15;
    const int half = lane >> 4;
    const int m0   = blockIdx.x * 256 + wave * 32;
    const int n0   = blockIdx.y * 64;

    v8f acc[2][4];
    for (int mi = 0; mi < 2; ++mi)
        for (int c = 0; c < 4; ++c)
            for (int i = 0; i < 8; ++i) acc[mi][c][i] = 0.f;

    const unsigned short* arow0 = A + (size_t)(m0 + nl) * K + half * 8;
    const unsigned short* arow1 = arow0 + (size_t)16 * K;
    const unsigned short* brow[4];
#pragma unroll
    for (int c = 0; c < 4; ++c)
        brow[c] = B + (size_t)(n0 + c * 16 + nl) * K + half * 8;

    Frag16 a0[2], b0[4], a1[2], b1[4];
    a0[0].q[0] = *(const uint4*)(arow0);
    a0[0].q[1] = *(const uint4*)(arow0 + 16);
    a0[1].q[0] = *(const uint4*)(arow1);
    a0[1].q[1] = *(const uint4*)(arow1 + 16);
#pragma unroll
    for (int c = 0; c < 4; ++c) {
        b0[c].q[0] = *(const uint4*)(brow[c]);
        b0[c].q[1] = *(const uint4*)(brow[c] + 16);
    }

    for (int k = 0; k < K; k += 64) {
        // prefetch k+32 into buffer 1
        a1[0].q[0] = *(const uint4*)(arow0 + k + 32);
        a1[0].q[1] = *(const uint4*)(arow0 + k + 48);
        a1[1].q[0] = *(const uint4*)(arow1 + k + 32);
        a1[1].q[1] = *(const uint4*)(arow1 + k + 48);
#pragma unroll
        for (int c = 0; c < 4; ++c) {
            b1[c].q[0] = *(const uint4*)(brow[c] + k + 32);
            b1[c].q[1] = *(const uint4*)(brow[c] + k + 48);
        }
        // consume buffer 0 (k)
#pragma unroll
        for (int c = 0; c < 4; ++c) {
            acc[0][c] = __builtin_amdgcn_wmma_f32_16x16x32_bf16(
                false, a0[0].v, false, b0[c].v, (short)0, acc[0][c], false, false);
            acc[1][c] = __builtin_amdgcn_wmma_f32_16x16x32_bf16(
                false, a0[1].v, false, b0[c].v, (short)0, acc[1][c], false, false);
        }
        // prefetch k+64 into buffer 0
        if (k + 64 < K) {
            a0[0].q[0] = *(const uint4*)(arow0 + k + 64);
            a0[0].q[1] = *(const uint4*)(arow0 + k + 80);
            a0[1].q[0] = *(const uint4*)(arow1 + k + 64);
            a0[1].q[1] = *(const uint4*)(arow1 + k + 80);
#pragma unroll
            for (int c = 0; c < 4; ++c) {
                b0[c].q[0] = *(const uint4*)(brow[c] + k + 64);
                b0[c].q[1] = *(const uint4*)(brow[c] + k + 80);
            }
        }
        // consume buffer 1 (k+32)
#pragma unroll
        for (int c = 0; c < 4; ++c) {
            acc[0][c] = __builtin_amdgcn_wmma_f32_16x16x32_bf16(
                false, a1[0].v, false, b1[c].v, (short)0, acc[0][c], false, false);
            acc[1][c] = __builtin_amdgcn_wmma_f32_16x16x32_bf16(
                false, a1[1].v, false, b1[c].v, (short)0, acc[1][c], false, false);
        }
    }
#pragma unroll
    for (int mi = 0; mi < 2; ++mi)
#pragma unroll
        for (int c = 0; c < 4; ++c)
#pragma unroll
            for (int r = 0; r < 8; ++r) {
                size_t off = (size_t)(m0 + mi * 16 + r + half * 8) * ldc +
                             (n0 + c * 16 + nl);
                if (BF16OUT) Cb[off] = f2bf(acc[mi][c][r]);
                else         Cf[off] = acc[mi][c][r];
            }
}

// ---------------------------------------------------------------------------
// RoPE in-place on bf16 [ntok, 1024] = [ntok, 16 heads, 64], with
// optional output scale (folds 1/sqrt(d_k) into Q).
// ---------------------------------------------------------------------------
__global__ void rope_bf16(unsigned short* __restrict__ t,
                          const int* __restrict__ pos, int npairs,
                          float oscale) {
    int idx = blockIdx.x * blockDim.x + threadIdx.x;
    if (idx >= npairs) return;
    int tok = idx >> 9;            // 512 pairs per token
    int p   = idx & 511;
    int h   = p >> 5;
    int i   = p & 31;
    int s   = tok & 2047;          // token = b*2048 + s
    float inv = __expf(-(float)(2 * i) * 0.015625f * 9.210340371976184f);
    float ang = (float)pos[s] * inv;
    float cs = cosf(ang) * oscale, sn = sinf(ang) * oscale;
    size_t base = (size_t)tok * 1024 + h * 64 + 2 * i;
    float x1 = bf2f(t[base]);
    float x2 = bf2f(t[base + 1]);
    t[base]     = f2bf(x1 * cs - x2 * sn);
    t[base + 1] = f2bf(x1 * sn + x2 * cs);
}

// ---------------------------------------------------------------------------
// Transpose V: [b,s,h,d] -> [b,h,d,s]  (bf16)
// ---------------------------------------------------------------------------
__global__ void transpose_v(const unsigned short* __restrict__ Vin,
                            unsigned short* __restrict__ Vt) {
    int idx = blockIdx.x * blockDim.x + threadIdx.x; // 4*16*64*2048
    int s  = idx & 2047;
    int d  = (idx >> 11) & 63;
    int bh = idx >> 17;
    int b = bh >> 4, h = bh & 15;
    Vt[idx] = Vin[((size_t)b * 2048 + s) * 1024 + h * 64 + d];
}

// ---------------------------------------------------------------------------
// One 32-key block of flash attention (DIAG = apply causal mask).
// ---------------------------------------------------------------------------
template <bool DIAG>
__device__ __forceinline__ void attn_kblock(
    int key0, int q0, int nl, int half,
    const unsigned short* __restrict__ Kp,
    const unsigned short* __restrict__ Vp,
    const Frag16 (&qbf)[2][2],
    unsigned short* __restrict__ Plds,        // [2][16*32]
    float (&m_i)[2], float (&l_i)[2], v8f (&acc)[2][4]) {
    constexpr int DM = 1024, S = 2048;
    // ---- S^T for both 16-key sub-tiles x both query tiles ----
    v8f st[2][2];   // [qtile][subtile]
#pragma unroll
    for (int j = 0; j < 2; ++j)
#pragma unroll
        for (int t = 0; t < 2; ++t)
            for (int i = 0; i < 8; ++i) st[j][t][i] = 0.f;
#pragma unroll
    for (int t = 0; t < 2; ++t) {
        const unsigned short* krow =
            Kp + (size_t)(key0 + t * 16 + nl) * DM + half * 8;
#pragma unroll
        for (int c2 = 0; c2 < 2; ++c2) {
            Frag16 kf;
            kf.q[0] = *(const uint4*)(krow + c2 * 32);
            kf.q[1] = *(const uint4*)(krow + c2 * 32 + 16);
#pragma unroll
            for (int j = 0; j < 2; ++j)
                st[j][t] = __builtin_amdgcn_wmma_f32_16x16x32_bf16(
                    false, kf.v, false, qbf[j][c2].v, (short)0,
                    st[j][t], false, false);
        }
    }
    // ---- causal mask: only the diagonal block pays for it ----
    if (DIAG) {
#pragma unroll
        for (int j = 0; j < 2; ++j) {
            const int q_lane = q0 + j * 16 + nl;
#pragma unroll
            for (int t = 0; t < 2; ++t)
#pragma unroll
                for (int r = 0; r < 8; ++r) {
                    const int key = key0 + t * 16 + half * 8 + r;
                    st[j][t][r] = (key <= q_lane) ? st[j][t][r] : -1e30f;
                }
        }
    }
    // ---- lane-local online softmax per query tile; P -> LDS ----
#pragma unroll
    for (int j = 0; j < 2; ++j) {
        float rm = st[j][0][0];
#pragma unroll
        for (int t = 0; t < 2; ++t)
#pragma unroll
            for (int r = 0; r < 8; ++r) rm = fmaxf(rm, st[j][t][r]);
        rm = fmaxf(rm, __shfl_xor(rm, 16, 32));   // combine key halves
        const float mn = fmaxf(m_i[j], rm);
        const float corr = __expf(m_i[j] - mn);
        m_i[j] = mn;
        float rs = 0.f;
#pragma unroll
        for (int t = 0; t < 2; ++t)
#pragma unroll
            for (int r = 0; r < 8; ++r) {
                st[j][t][r] = __expf(st[j][t][r] - mn);
                rs += st[j][t][r];
            }
        rs += __shfl_xor(rs, 16, 32);
        l_i[j] = l_i[j] * corr + rs;
#pragma unroll
        for (int c = 0; c < 4; ++c)
#pragma unroll
            for (int r = 0; r < 8; ++r) acc[j][c][r] *= corr;
#pragma unroll
        for (int t = 0; t < 2; ++t)
#pragma unroll
            for (int r = 0; r < 8; r += 2) {
                unsigned int pk = (unsigned int)f2bf(st[j][t][r]) |
                                  ((unsigned int)f2bf(st[j][t][r + 1]) << 16);
                *(unsigned int*)&Plds[j * 512 + nl * 32 + t * 16 + half * 8 + r] = pk;
            }
    }
    __syncthreads();
    // ---- O^T += V^T P : V loaded one tile at a time (low VGPR pressure) ----
    Frag16 pb[2];
#pragma unroll
    for (int j = 0; j < 2; ++j) {
        pb[j].q[0] = *(const uint4*)(&Plds[j * 512 + nl * 32 + half * 16]);
        pb[j].q[1] = *(const uint4*)(&Plds[j * 512 + nl * 32 + half * 16 + 8]);
    }
#pragma unroll
    for (int c = 0; c < 4; ++c) {
        Frag16 vf;
        const unsigned short* vrow =
            Vp + (size_t)(c * 16 + nl) * S + key0 + half * 16;
        vf.q[0] = *(const uint4*)(vrow);
        vf.q[1] = *(const uint4*)(vrow + 8);
#pragma unroll
        for (int j = 0; j < 2; ++j)
            acc[j][c] = __builtin_amdgcn_wmma_f32_16x16x32_bf16(
                false, vf.v, false, pb[j].v, (short)0, acc[j][c], false, false);
    }
    __syncthreads();
}

// ---------------------------------------------------------------------------
// Flash attention (causal), transposed formulation. 1 wave per block,
// 32 queries per wave (two 16-query tiles j=0,1 sharing K/V fragments).
// grid = (s/32 = 64, b*h = 64).
// S^T = K Q^T : lane owns query q0 + j*16 + lane%16; VGPR r owns key
//               key0 + t*16 + r + half*8.
// O^T = V^T P : acc[j][c][r] = O[d = c*16 + r + half*8][query_j].
// ---------------------------------------------------------------------------
__global__ __attribute__((amdgpu_flat_work_group_size(32, 32)))
void flash_attn(const unsigned short* __restrict__ Qb,
                const unsigned short* __restrict__ Kb,
                const unsigned short* __restrict__ Vt,
                unsigned short* __restrict__ Ob) {
    constexpr int S = 2048, DM = 1024, DK = 64;
    const int lane = threadIdx.x & 31;
    const int nl   = lane & 15;
    const int half = lane >> 4;
    const int qt = blockIdx.x;
    const int bh = blockIdx.y;
    const int b = bh >> 4, h = bh & 15;
    const int q0 = qt * 32;

    const unsigned short* Qp = Qb + (size_t)b * S * DM + h * DK;
    const unsigned short* Kp = Kb + (size_t)b * S * DM + h * DK;
    const unsigned short* Vp = Vt + (size_t)bh * DK * S;

    __shared__ __align__(16) unsigned short Plds[2 * 16 * 32]; // [qtile][q][key]

    // Q^T B-fragments: lane col = query q0 + j*16 + nl, table-order d.
    Frag16 qbf[2][2];
#pragma unroll
    for (int j = 0; j < 2; ++j) {
        const unsigned short* qrow =
            Qp + (size_t)(q0 + j * 16 + nl) * DM + half * 8;
#pragma unroll
        for (int c2 = 0; c2 < 2; ++c2) {
            qbf[j][c2].q[0] = *(const uint4*)(qrow + c2 * 32);
            qbf[j][c2].q[1] = *(const uint4*)(qrow + c2 * 32 + 16);
        }
    }

    float m_i[2] = {-1e30f, -1e30f}, l_i[2] = {0.f, 0.f};
    v8f acc[2][4];
    for (int j = 0; j < 2; ++j)
        for (int c = 0; c < 4; ++c)
            for (int i = 0; i < 8; ++i) acc[j][c][i] = 0.f;

    // kb = 0 .. qt-1 fully unmasked; kb = qt is the diagonal block.
    for (int kb = 0; kb < qt; ++kb)
        attn_kblock<false>(kb * 32, q0, nl, half, Kp, Vp, qbf, Plds,
                           m_i, l_i, acc);
    attn_kblock<true>(qt * 32, q0, nl, half, Kp, Vp, qbf, Plds,
                      m_i, l_i, acc);

    // ---- normalize, store O^T -> Ob[b, s=query, h*64+d] (16B stores) ----
#pragma unroll
    for (int j = 0; j < 2; ++j) {
        const float inv_l = 1.0f / l_i[j];
#pragma unroll
        for (int c = 0; c < 4; ++c) {
            Frag16 o8;                   // 8 bf16 = 16 bytes, contiguous in d
#pragma unroll
            for (int r = 0; r < 8; ++r) o8.u[r] = f2bf(acc[j][c][r] * inv_l);
            size_t off = ((size_t)b * S + q0 + j * 16 + nl) * DM + h * DK +
                         c * 16 + half * 8;
            *(uint4*)&Ob[off] = o8.q[0];
        }
    }
}

// ---------------------------------------------------------------------------
// Host-side launch
// ---------------------------------------------------------------------------
extern "C" void kernel_launch(void* const* d_in, const int* in_sizes, int n_in,
                              void* d_out, int out_size, void* d_ws,
                              size_t ws_size, hipStream_t stream) {
    const float* x  = (const float*)d_in[0];
    const float* Wq = (const float*)d_in[1];
    const float* Wk = (const float*)d_in[2];
    const float* Wv = (const float*)d_in[3];
    const float* Wo = (const float*)d_in[4];
    const int*   tp = (const int*)d_in[5];
    float* out = (float*)d_out;

    constexpr size_t NTOK = 4 * 2048;        // 8192 tokens
    constexpr size_t DM   = 1024;
    constexpr size_t XN   = NTOK * DM;       // 8,388,608
    constexpr size_t WN   = DM * DM;         // 1,048,576

    unsigned short* ws   = (unsigned short*)d_ws;
    unsigned short* xb   = ws;
    unsigned short* wqb  = xb + XN;
    unsigned short* wkb  = wqb + WN;
    unsigned short* wvb  = wkb + WN;
    unsigned short* wob  = wvb + WN;
    unsigned short* qbuf = wob + WN;
    unsigned short* kbuf = qbuf + XN;
    unsigned short* vtmp = kbuf + XN;
    unsigned short* vt   = vtmp + XN;
    unsigned short* ob   = vt + XN;

    // 1) convert inputs to bf16
    cvt_f32_bf16<<<(int)((XN + 255) / 256), 256, 0, stream>>>(x, xb, (int)XN);
    cvt_f32_bf16<<<(int)((WN + 255) / 256), 256, 0, stream>>>(Wq, wqb, (int)WN);
    cvt_f32_bf16<<<(int)((WN + 255) / 256), 256, 0, stream>>>(Wk, wkb, (int)WN);
    cvt_f32_bf16<<<(int)((WN + 255) / 256), 256, 0, stream>>>(Wv, wvb, (int)WN);
    cvt_f32_bf16<<<(int)((WN + 255) / 256), 256, 0, stream>>>(Wo, wob, (int)WN);

    // 2) Q/K/V projections: [8192,1024] = x @ W^T  (bf16 out)
    dim3 gg(32, 16);  // M/256, N/64
    gemm_bf16<true><<<gg, 256, 0, stream>>>(xb, wqb, nullptr, qbuf, 1024, 1024);
    gemm_bf16<true><<<gg, 256, 0, stream>>>(xb, wkb, nullptr, kbuf, 1024, 1024);
    gemm_bf16<true><<<gg, 256, 0, stream>>>(xb, wvb, nullptr, vtmp, 1024, 1024);

    // 3) RoPE on Q (scaled by 1/sqrt(d_k)) and K (in-place)
    const int npairs = (int)(NTOK * 512);
    rope_bf16<<<(npairs + 255) / 256, 256, 0, stream>>>(qbuf, tp, npairs, 0.125f);
    rope_bf16<<<(npairs + 255) / 256, 256, 0, stream>>>(kbuf, tp, npairs, 1.0f);

    // 4) transpose V per (b,h) -> [bh, 64, 2048]
    transpose_v<<<(int)(XN / 256), 256, 0, stream>>>(vtmp, vt);

    // 5) causal flash attention -> O (bf16, [b,s,1024])
    dim3 ag(64, 64);
    flash_attn<<<ag, 32, 0, stream>>>(qbuf, kbuf, vt, ob);

    // 6) output projection: out = O @ Wo^T  (f32 out)
    gemm_bf16<false><<<gg, 256, 0, stream>>>(ob, wob, out, nullptr, 1024, 1024);
}
